// AttentionBase_15968688406874
// MI455X (gfx1250) — matrix operand order repaired and checked by reading, not compiled, not run on hardware
//
#include <hip/hip_runtime.h>

// ---------------------------------------------------------------------------
// Attention block on gfx1250 (MI455X): all GEMMs via v_wmma_f32_16x16x32_f16,
// with tile staging done by the CDNA5 async global->LDS path
// (global_load_async_to_lds_b128, ASYNCcnt) and double-buffered LDS.
// C = A (MxK, f16 row-major) @ Bt^T (Bt is NxK f16 row-major), f32 accumulate.
// ---------------------------------------------------------------------------

typedef __attribute__((ext_vector_type(16))) _Float16 v16h;
typedef __attribute__((ext_vector_type(8)))  float    v8f;

union F16Frag { v16h v; uint4 q[2]; };

#define WG_THREADS 256
#define BM 256   // block rows (8 waves x 32 rows)
#define BN 64    // block cols
#define BK 32    // k-step (WMMA K)
#define LDSL 40  // padded LDS row stride in halves (conflict mitigation)

// Issue one K-step staging group: 4 async b128 for this thread's A row chunk
// (64 contiguous bytes), 1 async b128 for its B chunk.  ASYNCcnt += 5.
// NOTE: the instruction's offset: immediate applies to BOTH the global and the
// LDS address, and the LDS byte address is the low 32 bits of the generic ptr.
__device__ __forceinline__ void stage_async(unsigned adst, const _Float16* ag,
                                            unsigned bdst, const _Float16* bg)
{
    asm volatile(
        "global_load_async_to_lds_b128 %0, %2, off\n\t"
        "global_load_async_to_lds_b128 %0, %2, off offset:16\n\t"
        "global_load_async_to_lds_b128 %0, %2, off offset:32\n\t"
        "global_load_async_to_lds_b128 %0, %2, off offset:48\n\t"
        "global_load_async_to_lds_b128 %1, %3, off"
        :: "v"(adst), "v"(bdst), "v"(ag), "v"(bg)
        : "memory");
}

__device__ __forceinline__ void wait_async_le5() {
    asm volatile("s_wait_asynccnt 0x5" ::: "memory");
}
__device__ __forceinline__ void wait_async_0() {
    asm volatile("s_wait_asynccnt 0x0" ::: "memory");
}

__global__ __launch_bounds__(WG_THREADS)
void wmma_gemm_kernel(const _Float16* __restrict__ A,  unsigned long long sAb,
                      const _Float16* __restrict__ Bt, unsigned long long sBb,
                      void* __restrict__ outp,         unsigned long long sOb,
                      int out_is_f32,
                      const float* __restrict__ bias, int bias_mode, // 0 none, 1 col(bias[n]), 2 row(bias[m])
                      float scale, int N, int K)
{
    const int bz = blockIdx.z;
    A  += (size_t)bz * sAb;
    Bt += (size_t)bz * sBb;

    const int m0   = blockIdx.y * BM;
    const int n0   = blockIdx.x * BN;
    const int tid  = threadIdx.x;
    const int wave = tid >> 5;
    const int lane = tid & 31;
    const int lo   = lane & 15;
    const int hi   = lane >> 4;

    __shared__ _Float16 As[2][BM * LDSL];
    __shared__ _Float16 Bs[2][BN * LDSL];

    v8f acc[2][4];
#pragma unroll
    for (int mt = 0; mt < 2; ++mt)
#pragma unroll
        for (int t = 0; t < 4; ++t)
#pragma unroll
            for (int j = 0; j < 8; ++j) acc[mt][t][j] = 0.0f;

    // staging assignment: A = one 32-half row per thread (64B, 4 async b128)
    //                     B = 8 halves per thread (64 rows x 4 chunks)
    const int brow = tid >> 2, bchunk = tid & 3;
    const _Float16* Ag = A  + (size_t)(m0 + tid) * K;
    const _Float16* Bg = Bt + (size_t)(n0 + brow) * K + bchunk * 8;

    unsigned adst[2], bdst[2];
#pragma unroll
    for (int c = 0; c < 2; ++c) {
        adst[c] = (unsigned)(size_t)(const void*)&As[c][tid * LDSL];
        bdst[c] = (unsigned)(size_t)(const void*)&Bs[c][brow * LDSL + bchunk * 8];
    }

    // per-wave LDS fragment base offsets (half-indices)
    const int aoff0 = (wave * 32 + lo) * LDSL + hi * 8;
    const int aoff1 = (wave * 32 + 16 + lo) * LDSL + hi * 8;
    const int boff  = lo * LDSL + hi * 8;

    const int nsteps = K / BK;   // >= 32 for all uses
    stage_async(adst[0], Ag, bdst[0], Bg);

    // steady state: branch-free, double-buffered, async prefetch of next step
    for (int s = 0; s < nsteps - 1; ++s) {
        const int cur = s & 1;
        Ag += BK;
        Bg += BK;
        // prefetch next k-step into the other buffer (its last readers were
        // released by the barrier at the end of the previous step)
        stage_async(adst[cur ^ 1], Ag, bdst[cur ^ 1], Bg);
        wait_async_le5();       // older group (current buffer) retired
        __syncthreads();        // all waves' async data visible in LDS

        F16Frag af0, af1;
        {
            const _Float16* ap0 = &As[cur][aoff0];
            const _Float16* ap1 = &As[cur][aoff1];
            af0.q[0] = *reinterpret_cast<const uint4*>(ap0);
            af0.q[1] = *reinterpret_cast<const uint4*>(ap0 + 16);
            af1.q[0] = *reinterpret_cast<const uint4*>(ap1);
            af1.q[1] = *reinterpret_cast<const uint4*>(ap1 + 16);
        }
#pragma unroll
        for (int t = 0; t < 4; ++t) {
            const _Float16* bp = &Bs[cur][t * 16 * LDSL + boff];
            F16Frag bf;
            bf.q[0] = *reinterpret_cast<const uint4*>(bp);
            bf.q[1] = *reinterpret_cast<const uint4*>(bp + 16);
            acc[0][t] = __builtin_amdgcn_wmma_f32_16x16x32_f16(
                false, af0.v, false, bf.v, (short)0, acc[0][t], false, false);
            acc[1][t] = __builtin_amdgcn_wmma_f32_16x16x32_f16(
                false, af1.v, false, bf.v, (short)0, acc[1][t], false, false);
        }
        __syncthreads();        // reads done before next-step async overwrite
    }

    // peeled final k-step (no further staging)
    {
        const int cur = (nsteps - 1) & 1;
        wait_async_0();
        __syncthreads();

        F16Frag af0, af1;
        {
            const _Float16* ap0 = &As[cur][aoff0];
            const _Float16* ap1 = &As[cur][aoff1];
            af0.q[0] = *reinterpret_cast<const uint4*>(ap0);
            af0.q[1] = *reinterpret_cast<const uint4*>(ap0 + 16);
            af1.q[0] = *reinterpret_cast<const uint4*>(ap1);
            af1.q[1] = *reinterpret_cast<const uint4*>(ap1 + 16);
        }
#pragma unroll
        for (int t = 0; t < 4; ++t) {
            const _Float16* bp = &Bs[cur][t * 16 * LDSL + boff];
            F16Frag bf;
            bf.q[0] = *reinterpret_cast<const uint4*>(bp);
            bf.q[1] = *reinterpret_cast<const uint4*>(bp + 16);
            acc[0][t] = __builtin_amdgcn_wmma_f32_16x16x32_f16(
                false, af0.v, false, bf.v, (short)0, acc[0][t], false, false);
            acc[1][t] = __builtin_amdgcn_wmma_f32_16x16x32_f16(
                false, af1.v, false, bf.v, (short)0, acc[1][t], false, false);
        }
    }

    // epilogue: (acc + bias) * scale
    // C layout: element j -> m = mbase + hi*8 + j, n = n0 + t*16 + lo
#pragma unroll
    for (int mt = 0; mt < 2; ++mt) {
        const int mrow = m0 + wave * 32 + mt * 16 + hi * 8;
        float brow8[8];
        if (bias_mode == 2) {
#pragma unroll
            for (int j = 0; j < 8; ++j) brow8[j] = bias[mrow + j];
        }
#pragma unroll
        for (int t = 0; t < 4; ++t) {
            const int n = n0 + t * 16 + lo;
            const float bcol = (bias_mode == 1) ? bias[n] : 0.0f;
            if (out_is_f32) {
                float* O = reinterpret_cast<float*>(outp) + (size_t)bz * sOb;
#pragma unroll
                for (int j = 0; j < 8; ++j) {
                    float bv = (bias_mode == 2) ? brow8[j] : bcol;
                    O[(size_t)(mrow + j) * N + n] = (acc[mt][t][j] + bv) * scale;
                }
            } else {
                _Float16* O = reinterpret_cast<_Float16*>(outp) + (size_t)bz * sOb;
#pragma unroll
                for (int j = 0; j < 8; ++j) {
                    float bv = (bias_mode == 2) ? brow8[j] : bcol;
                    O[(size_t)(mrow + j) * N + n] =
                        (_Float16)((acc[mt][t][j] + bv) * scale);
                }
            }
        }
    }
}

// ---------------------------------------------------------------------------
// f32 -> f16 conversion (vectorized; n is a multiple of 4)
// ---------------------------------------------------------------------------
struct alignas(8) Half4 { _Float16 h[4]; };

__global__ __launch_bounds__(256)
void cvt_f32_to_f16_kernel(const float* __restrict__ src, _Float16* __restrict__ dst,
                           long long n)
{
    long long i = ((long long)blockIdx.x * 256 + threadIdx.x) * 4;
    if (i + 3 < n) {
        float4 f = *reinterpret_cast<const float4*>(src + i);
        Half4 o;
        o.h[0] = (_Float16)f.x; o.h[1] = (_Float16)f.y;
        o.h[2] = (_Float16)f.z; o.h[3] = (_Float16)f.w;
        *reinterpret_cast<Half4*>(dst + i) = o;
    }
}

// ---------------------------------------------------------------------------
// Row softmax: one 256-thread block per row of 2048 f16 logits -> f16 probs
// ---------------------------------------------------------------------------
__global__ __launch_bounds__(256)
void softmax_kernel(const _Float16* __restrict__ L, _Float16* __restrict__ P, int cols)
{
    __shared__ float red[8];
    __shared__ float bcast;

    const int row = blockIdx.x;
    const int tid = threadIdx.x;

    union { uint4 q; _Float16 h[8]; } u;
    u.q = *reinterpret_cast<const uint4*>(L + (size_t)row * cols + tid * 8);

    float x[8];
    float mx = -3.0e38f;
#pragma unroll
    for (int i = 0; i < 8; ++i) { x[i] = (float)u.h[i]; mx = fmaxf(mx, x[i]); }

    for (int o = 16; o > 0; o >>= 1) mx = fmaxf(mx, __shfl_xor(mx, o, 32));
    if ((tid & 31) == 0) red[tid >> 5] = mx;
    __syncthreads();
    if (tid == 0) {
        float m = red[0];
        for (int i = 1; i < 8; ++i) m = fmaxf(m, red[i]);
        bcast = m;
    }
    __syncthreads();
    mx = bcast;

    float s = 0.0f;
#pragma unroll
    for (int i = 0; i < 8; ++i) { x[i] = __expf(x[i] - mx); s += x[i]; }
    for (int o = 16; o > 0; o >>= 1) s += __shfl_xor(s, o, 32);
    __syncthreads();  // everyone has consumed bcast/red
    if ((tid & 31) == 0) red[tid >> 5] = s;
    __syncthreads();
    if (tid == 0) {
        float t = 0.0f;
        for (int i = 0; i < 8; ++i) t += red[i];
        bcast = t;
    }
    __syncthreads();
    const float inv = 1.0f / bcast;

    union { uint4 q; _Float16 h[8]; } o;
#pragma unroll
    for (int i = 0; i < 8; ++i) o.h[i] = (_Float16)(x[i] * inv);
    *reinterpret_cast<uint4*>(P + (size_t)row * cols + tid * 8) = o.q;
}

// ---------------------------------------------------------------------------
// Host launch
// ---------------------------------------------------------------------------
extern "C" void kernel_launch(void* const* d_in, const int* in_sizes, int n_in,
                              void* d_out, int out_size, void* d_ws, size_t ws_size,
                              hipStream_t stream)
{
    (void)in_sizes; (void)n_in; (void)out_size; (void)ws_size;
    const int Bb = 8, S = 2048, C = 1024;
    typedef unsigned long long u64;

    const float* input  = (const float*)d_in[0];
    const float* memory = (const float*)d_in[1];
    const float* Wq = (const float*)d_in[2];
    const float* bq = (const float*)d_in[3];
    const float* Wk = (const float*)d_in[4];
    const float* bk = (const float*)d_in[5];
    const float* Wv = (const float*)d_in[6];
    const float* bv = (const float*)d_in[7];
    const float* Wo = (const float*)d_in[8];
    const float* bo = (const float*)d_in[9];
    float* out = (float*)d_out;

    // Workspace layout (peak 168 MB) with lifetime-based reuse:
    //  [  0.. 32) inF16    -> later L16 (0..64) -> later sel16 (0..32)
    //  [ 32.. 64) memF16   -> later part of L16
    //  [ 64.. 72) Wq/Wk/Wv/Wo f16 (2 MB each)
    //  [ 72..104) Q16      -> later part of P16 (72..136)
    //  [104..136) K16      -> later part of P16
    //  [136..168) VT16
    char* ws = (char*)d_ws;
    const size_t MB = 1ull << 20;
    _Float16* inF16  = (_Float16*)(ws +   0 * MB);
    _Float16* memF16 = (_Float16*)(ws +  32 * MB);
    _Float16* Wq16   = (_Float16*)(ws +  64 * MB);
    _Float16* Wk16   = (_Float16*)(ws +  66 * MB);
    _Float16* Wv16   = (_Float16*)(ws +  68 * MB);
    _Float16* Wo16   = (_Float16*)(ws +  70 * MB);
    _Float16* Q16    = (_Float16*)(ws +  72 * MB);
    _Float16* K16    = (_Float16*)(ws + 104 * MB);
    _Float16* VT16   = (_Float16*)(ws + 136 * MB);
    _Float16* L16    = (_Float16*)(ws +   0 * MB);  // reuse: after VT gemm
    _Float16* P16    = (_Float16*)(ws +  72 * MB);  // reuse: after logits
    _Float16* S16    = (_Float16*)(ws +   0 * MB);  // reuse: after softmax

    const float depth_scale = 0.03125f;  // 1024^-0.5

    // 1) f32 -> f16 conversions
    long long nIn = (long long)Bb * S * C;   // 16,777,216
    long long nW  = (long long)C * C;        //  1,048,576
    cvt_f32_to_f16_kernel<<<(unsigned)(nIn / 1024), 256, 0, stream>>>(input,  inF16,  nIn);
    cvt_f32_to_f16_kernel<<<(unsigned)(nIn / 1024), 256, 0, stream>>>(memory, memF16, nIn);
    cvt_f32_to_f16_kernel<<<(unsigned)(nW  / 1024), 256, 0, stream>>>(Wq, Wq16, nW);
    cvt_f32_to_f16_kernel<<<(unsigned)(nW  / 1024), 256, 0, stream>>>(Wk, Wk16, nW);
    cvt_f32_to_f16_kernel<<<(unsigned)(nW  / 1024), 256, 0, stream>>>(Wv, Wv16, nW);
    cvt_f32_to_f16_kernel<<<(unsigned)(nW  / 1024), 256, 0, stream>>>(Wo, Wo16, nW);

    // 2) Q = (input @ Wq^T + bq) * depth_scale         [16384 x 1024] f16
    wmma_gemm_kernel<<<dim3(C / BN, (Bb * S) / BM, 1), WG_THREADS, 0, stream>>>(
        inF16, 0ull, Wq16, 0ull, Q16, 0ull, 0, bq, 1, depth_scale, C, C);

    // 3) K = memory @ Wk^T + bk                        [16384 x 1024] f16
    wmma_gemm_kernel<<<dim3(C / BN, (Bb * S) / BM, 1), WG_THREADS, 0, stream>>>(
        memF16, 0ull, Wk16, 0ull, K16, 0ull, 0, bk, 1, 1.0f, C, C);

    // 4) V^T = Wv @ memory^T + bv (row bias)  per batch [1024 x 2048] f16
    wmma_gemm_kernel<<<dim3(S / BN, C / BM, Bb), WG_THREADS, 0, stream>>>(
        Wv16, 0ull, memF16, (u64)S * C, VT16, (u64)C * S, 0, bv, 2, 1.0f, S, C);

    // 5) logits = Q @ K^T          per batch [2048 x 2048] f16
    wmma_gemm_kernel<<<dim3(S / BN, S / BM, Bb), WG_THREADS, 0, stream>>>(
        Q16, (u64)S * C, K16, (u64)S * C, L16, (u64)S * S, 0, nullptr, 0, 1.0f, S, C);

    // 6) P = softmax(logits)       per row of 2048
    softmax_kernel<<<Bb * S, 256, 0, stream>>>(L16, P16, S);

    // 7) sel = P @ (V^T)^T         per batch [2048 x 1024] f16
    wmma_gemm_kernel<<<dim3(C / BN, S / BM, Bb), WG_THREADS, 0, stream>>>(
        P16, (u64)S * S, VT16, (u64)C * S, S16, (u64)S * C, 0, nullptr, 0, 1.0f, C, S);

    // 8) out = sel @ Wo^T + bo     [16384 x 1024] f32
    wmma_gemm_kernel<<<dim3(C / BN, (Bb * S) / BM, 1), WG_THREADS, 0, stream>>>(
        S16, 0ull, Wo16, 0ull, out, 0ull, 1, bo, 1, 1.0f, C, C);
}
